// LinearAttention_29033978921077
// MI455X (gfx1250) — matrix operand order repaired and checked by reading, not compiled, not run on hardware
//
#include <hip/hip_runtime.h>
#include <hip/hip_bf16.h>

typedef __attribute__((ext_vector_type(16))) _Float16 v16h;
typedef __attribute__((ext_vector_type(8)))  float    v8f;

#define N_      8
#define L_      4096
#define S_      4096
#define H_      8
#define D_      64
#define M_      64
#define NH_     (N_ * H_)          // 64
#define EPS_    1e-6f

#define SCHUNKS 8                  // S split across blocks per head
#define SCHUNK  (S_ / SCHUNKS)     // 512
#define STILE   64                 // s rows staged per iteration
#define LTILE   64                 // L rows per phase-2 block
#define LDSP    72                 // padded LDS row stride (halves)
#define PART    (D_ * M_ + D_)     // 4160 floats: KV[64][64] + Ksum[64]

__device__ __forceinline__ float phi_elu1(float x) {
    // elu(x) + 1 == x+1 (x>0) else exp(x)
    return x > 0.0f ? x + 1.0f : __expf(x);
}

// A-matrix fragment, 16x32 f16, from LDS laid out [row][k] with given stride.
// Lane L<16 -> M=L, K=kc+0..15; lanes 16-31 -> M=L-16, K=kc+16..31.
__device__ __forceinline__ v16h load_frag_a(const _Float16* lds, int stride,
                                            int r0, int kc, int lane) {
    v16h a;
    const _Float16* p = lds + (r0 + (lane & 15)) * stride;
    int kbase = kc + ((lane >> 4) << 4);
#pragma unroll
    for (int i = 0; i < 16; ++i) {
        int k = kbase + i;
        a[i] = p[k];
    }
    return a;
}

// B-matrix fragment, 32x16 f16, from LDS laid out [n][k] with given stride.
// Lane L<16 -> N=L, K=kc+0..15; lanes 16-31 -> N=L-16, K=kc+16..31.
__device__ __forceinline__ v16h load_frag_b(const _Float16* lds, int stride,
                                            int c0, int kc, int lane) {
    v16h b;
    const _Float16* p = lds + (c0 + (lane & 15)) * stride + kc + ((lane >> 4) << 4);
#pragma unroll
    for (int i = 0; i < 16; ++i) b[i] = p[i];
    return b;
}

__device__ __forceinline__ v8f wmma_f16(v16h a, v16h b, v8f c) {
    return __builtin_amdgcn_wmma_f32_16x16x32_f16(
        /*neg_a=*/false, a, /*neg_b=*/false, b,
        /*c_mod=*/(short)0, c, /*reuse_a=*/false, /*reuse_b=*/false);
}

// ---------------- Phase 1: partial KV[m][d] = sum_s V[s,m]*phiK[s,d], + Ksum ----
__global__ __launch_bounds__(128) void la_phase1(const float* __restrict__ keys,
                                                 const float* __restrict__ values,
                                                 const float* __restrict__ key_mask,
                                                 float* __restrict__ part) {
    __shared__ _Float16 vT[M_ * LDSP];      // vT[m][s_local] = V[s,m]
    __shared__ _Float16 kT[D_ * LDSP];      // kT[d][s_local] = phiK[s,d]
    __shared__ float    ksp[128 * 32];      // ksum partials (fixed-order reduce)

    const int t    = threadIdx.x;
    const int lane = t & 31;
    const int wv   = t >> 5;
    const int nh   = blockIdx.x >> 3;
    const int ck   = blockIdx.x & 7;
    const int n    = nh >> 3;
    const int h    = nh & 7;
    const int s0   = ck * SCHUNK;

    const int srow  = t >> 1;      // s index within tile (0..63)
    const int halfd = t & 1;       // which 32-wide half of D/M

    float kacc[32];
#pragma unroll
    for (int i = 0; i < 32; ++i) kacc[i] = 0.0f;

    v8f acc[4];
#pragma unroll
    for (int cb = 0; cb < 4; ++cb)
#pragma unroll
        for (int e = 0; e < 8; ++e) acc[cb][e] = 0.0f;

    for (int tile = 0; tile < SCHUNK / STILE; ++tile) {
        __syncthreads();
        const int s = s0 + tile * STILE + srow;
        const float* kp = keys   + ((size_t)((size_t)n * S_ + s) * H_ + h) * D_ + halfd * 32;
        const float* vp = values + ((size_t)((size_t)n * S_ + s) * H_ + h) * M_ + halfd * 32;
        const float  mk = key_mask[(size_t)n * S_ + s];
#pragma unroll
        for (int q = 0; q < 8; ++q) {
            float4 k4 = ((const float4*)kp)[q];
            float4 v4 = ((const float4*)vp)[q];
            float p0 = phi_elu1(k4.x) * mk;
            float p1 = phi_elu1(k4.y) * mk;
            float p2 = phi_elu1(k4.z) * mk;
            float p3 = phi_elu1(k4.w) * mk;
            kacc[q * 4 + 0] += p0; kacc[q * 4 + 1] += p1;
            kacc[q * 4 + 2] += p2; kacc[q * 4 + 3] += p3;
            const int db = halfd * 32 + q * 4;
            kT[(db + 0) * LDSP + srow] = (_Float16)p0;
            kT[(db + 1) * LDSP + srow] = (_Float16)p1;
            kT[(db + 2) * LDSP + srow] = (_Float16)p2;
            kT[(db + 3) * LDSP + srow] = (_Float16)p3;
            vT[(db + 0) * LDSP + srow] = (_Float16)v4.x;
            vT[(db + 1) * LDSP + srow] = (_Float16)v4.y;
            vT[(db + 2) * LDSP + srow] = (_Float16)v4.z;
            vT[(db + 3) * LDSP + srow] = (_Float16)v4.w;
        }
        __syncthreads();

        v16h a0 = load_frag_a(vT, LDSP, wv * 16, 0,  lane);
        v16h a1 = load_frag_a(vT, LDSP, wv * 16, 32, lane);
#pragma unroll
        for (int cb = 0; cb < 4; ++cb) {
            v16h b0 = load_frag_b(kT, LDSP, cb * 16, 0,  lane);
            v16h b1 = load_frag_b(kT, LDSP, cb * 16, 32, lane);
            acc[cb] = wmma_f16(a0, b0, acc[cb]);
            acc[cb] = wmma_f16(a1, b1, acc[cb]);
        }
    }

    // write private KV partial (every element exactly once -> no init needed)
    float* kvout = part + (size_t)blockIdx.x * PART;
#pragma unroll
    for (int cb = 0; cb < 4; ++cb)
#pragma unroll
        for (int j = 0; j < 8; ++j) {
            int m = wv * 16 + j + ((lane >> 4) << 3);
            int d = cb * 16 + (lane & 15);
            kvout[m * D_ + d] = acc[cb][j];
        }

    // deterministic Ksum reduction (fixed summation order)
#pragma unroll
    for (int i = 0; i < 32; ++i) ksp[t * 32 + i] = kacc[i];
    __syncthreads();
    if (t < 64) {
        const int d  = t;
        const int hb = d >> 5;
        float s = 0.0f;
        for (int g = 0; g < 64; ++g) s += ksp[(2 * g + hb) * 32 + (d & 31)];
        kvout[D_ * M_ + d] = s;
    }
}

// ---------------- Reduce: sum 8 chunk-partials per head, fixed order ----------
__global__ void la_reduce(const float* __restrict__ part, float* __restrict__ red,
                          int total) {
    int i = blockIdx.x * blockDim.x + threadIdx.x;
    if (i >= total) return;
    int nh  = i / PART;
    int off = i - nh * PART;
    float s = 0.0f;
#pragma unroll
    for (int c = 0; c < SCHUNKS; ++c)
        s += part[((size_t)nh * SCHUNKS + c) * PART + off];
    red[i] = s;
}

// ---------------- Phase 2: out[l,m] = (phiQ[l,:]·KV[m,:]) * 1/(phiQ·Ksum+eps) --
__global__ __launch_bounds__(128) void la_phase2(const float* __restrict__ queries,
                                                 const float* __restrict__ red,
                                                 float* __restrict__ out) {
    __shared__ _Float16 qh[LTILE * LDSP];   // qh[l_local][d] = phiQ
    __shared__ _Float16 kvh[M_ * LDSP];     // kvh[m][d] = KV[m,d] (acts as B=KV^T)
    __shared__ float    zpart[64 * 2];
    __shared__ float    zinv[64];

    const int t    = threadIdx.x;
    const int lane = t & 31;
    const int wv   = t >> 5;
    const int nh   = blockIdx.x >> 6;
    const int lt   = blockIdx.x & 63;
    const int n    = nh >> 3;
    const int h    = nh & 7;
    const int l0   = lt * LTILE;

    const float* redp = red + (size_t)nh * PART;
    const int row   = t >> 1;
    const int halfd = t & 1;

    // stage KV as f16
    {
        const float* kvp = redp + row * D_ + halfd * 32;
#pragma unroll
        for (int q = 0; q < 8; ++q) {
            float4 v4 = ((const float4*)kvp)[q];
            const int db = halfd * 32 + q * 4;
            kvh[row * LDSP + db + 0] = (_Float16)v4.x;
            kvh[row * LDSP + db + 1] = (_Float16)v4.y;
            kvh[row * LDSP + db + 2] = (_Float16)v4.z;
            kvh[row * LDSP + db + 3] = (_Float16)v4.w;
        }
    }

    // stage phi(Q) tile as f16 + accumulate Z partial against Ksum
    float zp = 0.0f;
    {
        const int l = l0 + row;
        const float* qp = queries + ((size_t)((size_t)n * L_ + l) * H_ + h) * D_ + halfd * 32;
        const float* ks = redp + D_ * M_ + halfd * 32;
#pragma unroll
        for (int q = 0; q < 8; ++q) {
            float4 q4 = ((const float4*)qp)[q];
            float4 k4 = ((const float4*)ks)[q];
            float p0 = phi_elu1(q4.x);
            float p1 = phi_elu1(q4.y);
            float p2 = phi_elu1(q4.z);
            float p3 = phi_elu1(q4.w);
            zp += p0 * k4.x + p1 * k4.y + p2 * k4.z + p3 * k4.w;
            const int db = halfd * 32 + q * 4;
            qh[row * LDSP + db + 0] = (_Float16)p0;
            qh[row * LDSP + db + 1] = (_Float16)p1;
            qh[row * LDSP + db + 2] = (_Float16)p2;
            qh[row * LDSP + db + 3] = (_Float16)p3;
        }
    }
    zpart[row * 2 + halfd] = zp;
    __syncthreads();
    if (t < 64) zinv[t] = 1.0f / (zpart[t * 2] + zpart[t * 2 + 1] + EPS_);
    __syncthreads();

    v16h a0 = load_frag_a(qh, LDSP, wv * 16, 0,  lane);
    v16h a1 = load_frag_a(qh, LDSP, wv * 16, 32, lane);
#pragma unroll
    for (int cb = 0; cb < 4; ++cb) {
        v16h b0 = load_frag_b(kvh, LDSP, cb * 16, 0,  lane);
        v16h b1 = load_frag_b(kvh, LDSP, cb * 16, 32, lane);
        v8f c = {};
        c = wmma_f16(a0, b0, c);
        c = wmma_f16(a1, b1, c);
#pragma unroll
        for (int j = 0; j < 8; ++j) {
            int ml  = wv * 16 + j + ((lane >> 4) << 3);
            int col = cb * 16 + (lane & 15);
            out[((size_t)((size_t)n * L_ + l0 + ml) * H_ + h) * M_ + col] = c[j] * zinv[ml];
        }
    }
}

extern "C" void kernel_launch(void* const* d_in, const int* in_sizes, int n_in,
                              void* d_out, int out_size, void* d_ws, size_t ws_size,
                              hipStream_t stream) {
    const float* queries  = (const float*)d_in[0];
    const float* keys     = (const float*)d_in[1];
    const float* values   = (const float*)d_in[2];
    const float* key_mask = (const float*)d_in[3];
    float* out = (float*)d_out;

    float* part = (float*)d_ws;                                   // [NH*SCHUNKS][PART]
    float* red  = part + (size_t)NH_ * SCHUNKS * PART;            // [NH][PART]

    la_phase1<<<dim3(NH_ * SCHUNKS), dim3(128), 0, stream>>>(keys, values, key_mask, part);

    const int total = NH_ * PART;
    la_reduce<<<dim3((total + 255) / 256), dim3(256), 0, stream>>>(part, red, total);

    la_phase2<<<dim3(NH_ * (L_ / LTILE)), dim3(128), 0, stream>>>(queries, red, out);
}
